// MultiheadAttentionSeparatedLiT_23381801959857
// MI455X (gfx1250) — compile-verified, matrix operand order checked by reading, not run on hardware
//
#include <hip/hip_runtime.h>
#include <hip/hip_bf16.h>
#include <stdint.h>
#include <math.h>

// Problem dims (fixed by the reference)
#define S_LEN 2048
#define BATCH 4
#define EMB   1024
#define HEADS 16
#define HDIM  64

typedef __attribute__((ext_vector_type(16))) _Float16 v16h;
typedef __attribute__((ext_vector_type(8)))  float    v8f;

union HFrag { v16h h; uint32_t u[8]; };

// ---------------------------------------------------------------------------
// Fragment loaders (16x16x32 f16 WMMA, wave32)
// A-layout (ISA 7.12.2, 16-bit A 16x32): lane L holds row M=L&15;
//   K base = 8*(L>=16); VGPR j<4 -> K pair (2j, 2j+1); j>=4 -> (16+2(j-4), ...).
// B operand is always supplied as an [N x K] row-major matrix (i.e. already
// transposed), loaded with the symmetric layout: lane -> column N, same K walk.
// ---------------------------------------------------------------------------
__device__ __forceinline__ v16h load_frag_h(const _Float16* __restrict__ base,
                                            int ld, int row0, int col0, int lane) {
  const _Float16* p = base + (size_t)(row0 + (lane & 15)) * ld + col0 + ((lane >> 4) << 3);
  HFrag f;
#pragma unroll
  for (int j = 0; j < 8; ++j) {
    const int off = (j < 4) ? (2 * j) : (16 + 2 * (j - 4));
    f.u[j] = *(const uint32_t*)(p + off);
  }
  return f.h;
}

// Same fragment but gathering from an f32 matrix with in-register f16 convert
// (used for the softmaxed attention probabilities).
__device__ __forceinline__ v16h load_frag_f32(const float* __restrict__ base,
                                              int ld, int row0, int col0, int lane) {
  const float* p = base + (size_t)(row0 + (lane & 15)) * ld + col0 + ((lane >> 4) << 3);
  HFrag f;
#pragma unroll
  for (int j = 0; j < 8; ++j) {
    const int off = (j < 4) ? (2 * j) : (16 + 2 * (j - 4));
    f.h[2 * j]     = (_Float16)p[off];
    f.h[2 * j + 1] = (_Float16)p[off + 1];
  }
  return f.h;
}

__device__ __forceinline__ v8f wmma16(v16h a, v16h b, v8f c) {
  // D = A x B + C ; f32 accumulate
  return __builtin_amdgcn_wmma_f32_16x16x32_f16(false, a, false, b, (short)0, c,
                                                false, false);
}

// ---------------------------------------------------------------------------
// f32 -> f16 conversion
// ---------------------------------------------------------------------------
__global__ void cvt_f16_kernel(const float* __restrict__ in,
                               _Float16* __restrict__ out, int n) {
  int i = blockIdx.x * blockDim.x + threadIdx.x;
  if (i < n) out[i] = (_Float16)in[i];
}

// ---------------------------------------------------------------------------
// In-projection: P = X @ W^T + bias
//   X  : f16 [M=S*B x K=E], row m = s*BATCH + b (native [S,B,E] flattening)
//   W  : f16 [N=E x K=E] row-major (so W^T multiply == B-fragment from [N,K])
//   store_mode 0: out f16 at ((b*H+h)*S + s)*D + d      (Q, K)
//   store_mode 1: out f16 at ((b*H+h)*D + d)*S + s      (V, pre-transposed)
// Block: 128 threads = 4 waves, 64x64 tile; wave = 32x32 (2x2 fragments).
// ---------------------------------------------------------------------------
__global__ void proj_qkv_kernel(const _Float16* __restrict__ X,
                                const _Float16* __restrict__ W,
                                const float* __restrict__ bias,
                                _Float16* __restrict__ out, int store_mode) {
  const int lane  = threadIdx.x & 31;
  const int wave  = threadIdx.x >> 5;
  const int waveM = blockIdx.x * 64 + (wave >> 1) * 32;
  const int waveN = blockIdx.y * 64 + (wave & 1) * 32;

  v8f c[2][2] = {};
  for (int k0 = 0; k0 < EMB; k0 += 32) {
    v16h a0 = load_frag_h(X, EMB, waveM,      k0, lane);
    v16h a1 = load_frag_h(X, EMB, waveM + 16, k0, lane);
    v16h b0 = load_frag_h(W, EMB, waveN,      k0, lane);
    v16h b1 = load_frag_h(W, EMB, waveN + 16, k0, lane);
    c[0][0] = wmma16(a0, b0, c[0][0]);
    c[0][1] = wmma16(a0, b1, c[0][1]);
    c[1][0] = wmma16(a1, b0, c[1][0]);
    c[1][1] = wmma16(a1, b1, c[1][1]);
  }

#pragma unroll
  for (int mi = 0; mi < 2; ++mi)
#pragma unroll
    for (int ni = 0; ni < 2; ++ni)
#pragma unroll
      for (int r = 0; r < 8; ++r) {
        const int m = waveM + mi * 16 + r + ((lane >> 4) << 3);
        const int n = waveN + ni * 16 + (lane & 15);
        const float val = c[mi][ni][r] + bias[n];
        const int s = m / BATCH, b = m % BATCH;
        const int h = n / HDIM,  d = n % HDIM;
        const size_t addr = (store_mode == 0)
            ? ((size_t)(b * HEADS + h) * S_LEN + s) * HDIM + d
            : ((size_t)(b * HEADS + h) * HDIM + d) * S_LEN + s;
        out[addr] = (_Float16)val;
      }
}

// ---------------------------------------------------------------------------
// Logits: attn[bh, q, k] = (Q[bh] . K[bh]) * 1/sqrt(D)   (f32 to d_out)
// Q,K f16 [B*H, S, D].  M = N = S, K-dim = D (2 WMMA k-steps).
// ---------------------------------------------------------------------------
__global__ void qk_logits_kernel(const _Float16* __restrict__ Qp,
                                 const _Float16* __restrict__ Kp,
                                 float* __restrict__ attn) {
  const int lane  = threadIdx.x & 31;
  const int wave  = threadIdx.x >> 5;
  const int bh    = blockIdx.z;
  const int waveM = blockIdx.x * 64 + (wave >> 1) * 32;
  const int waveN = blockIdx.y * 64 + (wave & 1) * 32;

  const _Float16* Q = Qp + (size_t)bh * S_LEN * HDIM;
  const _Float16* K = Kp + (size_t)bh * S_LEN * HDIM;
  float* A = attn + (size_t)bh * S_LEN * S_LEN;

  v8f c[2][2] = {};
#pragma unroll
  for (int k0 = 0; k0 < HDIM; k0 += 32) {
    v16h a0 = load_frag_h(Q, HDIM, waveM,      k0, lane);
    v16h a1 = load_frag_h(Q, HDIM, waveM + 16, k0, lane);
    v16h b0 = load_frag_h(K, HDIM, waveN,      k0, lane);   // B[k][n] = K[n][k]
    v16h b1 = load_frag_h(K, HDIM, waveN + 16, k0, lane);
    c[0][0] = wmma16(a0, b0, c[0][0]);
    c[0][1] = wmma16(a0, b1, c[0][1]);
    c[1][0] = wmma16(a1, b0, c[1][0]);
    c[1][1] = wmma16(a1, b1, c[1][1]);
  }

  const float scale = 0.125f;  // 1/sqrt(64)
#pragma unroll
  for (int mi = 0; mi < 2; ++mi)
#pragma unroll
    for (int ni = 0; ni < 2; ++ni)
#pragma unroll
      for (int r = 0; r < 8; ++r) {
        const int m = waveM + mi * 16 + r + ((lane >> 4) << 3);
        const int n = waveN + ni * 16 + (lane & 15);
        A[(size_t)m * S_LEN + n] = c[mi][ni][r] * scale;
      }
}

// ---------------------------------------------------------------------------
// Row softmax in place over d_out attention region. One block = one row of
// 2048 logits; 256 threads x 8 elements; LDS tree reductions.
// ---------------------------------------------------------------------------
__global__ void softmax_kernel(float* __restrict__ attn) {
  __shared__ float red[256];
  const int t = threadIdx.x;
  float* row = attn + (size_t)blockIdx.x * S_LEN;

  float v[8];
  float m = -INFINITY;
#pragma unroll
  for (int i = 0; i < 8; ++i) { v[i] = row[t + i * 256]; m = fmaxf(m, v[i]); }
  red[t] = m; __syncthreads();
  for (int off = 128; off > 0; off >>= 1) {
    if (t < off) red[t] = fmaxf(red[t], red[t + off]);
    __syncthreads();
  }
  m = red[0]; __syncthreads();

  float s = 0.f;
#pragma unroll
  for (int i = 0; i < 8; ++i) { v[i] = __expf(v[i] - m); s += v[i]; }
  red[t] = s; __syncthreads();
  for (int off = 128; off > 0; off >>= 1) {
    if (t < off) red[t] += red[t + off];
    __syncthreads();
  }
  const float inv = 1.0f / red[0];
#pragma unroll
  for (int i = 0; i < 8; ++i) row[t + i * 256] = v[i] * inv;
}

// ---------------------------------------------------------------------------
// AV: Y[bh] = P[bh] @ V[bh].  P f32 [S x S] (converted in-register to f16),
// Vt f16 [B*H, D, S] (pre-transposed so B-fragment gathers are contiguous).
// Output Y f16 in [B, S, E] layout: (b*S+s)*E + h*D + d.
// ---------------------------------------------------------------------------
__global__ void av_kernel(const float* __restrict__ attn,
                          const _Float16* __restrict__ Vt,
                          _Float16* __restrict__ Y) {
  const int lane  = threadIdx.x & 31;
  const int wave  = threadIdx.x >> 5;
  const int bh    = blockIdx.z;
  const int waveM = blockIdx.x * 64 + (wave >> 1) * 32;
  const int waveN = (wave & 1) * 32;           // N = D = 64 total

  const float* P = attn + (size_t)bh * S_LEN * S_LEN;
  const _Float16* VT = Vt + (size_t)bh * HDIM * S_LEN;

  v8f c[2][2] = {};
  for (int k0 = 0; k0 < S_LEN; k0 += 32) {
    v16h a0 = load_frag_f32(P, S_LEN, waveM,      k0, lane);
    v16h a1 = load_frag_f32(P, S_LEN, waveM + 16, k0, lane);
    v16h b0 = load_frag_h(VT, S_LEN, waveN,      k0, lane);  // B[k][n] = Vt[n][k]
    v16h b1 = load_frag_h(VT, S_LEN, waveN + 16, k0, lane);
    c[0][0] = wmma16(a0, b0, c[0][0]);
    c[0][1] = wmma16(a0, b1, c[0][1]);
    c[1][0] = wmma16(a1, b0, c[1][0]);
    c[1][1] = wmma16(a1, b1, c[1][1]);
  }

  const int b = bh / HEADS, h = bh % HEADS;
#pragma unroll
  for (int mi = 0; mi < 2; ++mi)
#pragma unroll
    for (int ni = 0; ni < 2; ++ni)
#pragma unroll
      for (int r = 0; r < 8; ++r) {
        const int s = waveM + mi * 16 + r + ((lane >> 4) << 3);
        const int d = waveN + ni * 16 + (lane & 15);
        Y[((size_t)b * S_LEN + s) * EMB + h * HDIM + d] = (_Float16)c[mi][ni][r];
      }
}

// ---------------------------------------------------------------------------
// Out-projection: out = Y @ Wo^T + bo, fused [B,S,E] -> [S,B,E] transpose,
// f32 store straight into d_out region 0.
// ---------------------------------------------------------------------------
__global__ void out_proj_kernel(const _Float16* __restrict__ Y,
                                const _Float16* __restrict__ Wo,
                                const float* __restrict__ bo,
                                float* __restrict__ out) {
  const int lane  = threadIdx.x & 31;
  const int wave  = threadIdx.x >> 5;
  const int waveM = blockIdx.x * 64 + (wave >> 1) * 32;  // m = b*S + s
  const int waveN = blockIdx.y * 64 + (wave & 1) * 32;

  v8f c[2][2] = {};
  for (int k0 = 0; k0 < EMB; k0 += 32) {
    v16h a0 = load_frag_h(Y,  EMB, waveM,      k0, lane);
    v16h a1 = load_frag_h(Y,  EMB, waveM + 16, k0, lane);
    v16h b0 = load_frag_h(Wo, EMB, waveN,      k0, lane);
    v16h b1 = load_frag_h(Wo, EMB, waveN + 16, k0, lane);
    c[0][0] = wmma16(a0, b0, c[0][0]);
    c[0][1] = wmma16(a0, b1, c[0][1]);
    c[1][0] = wmma16(a1, b0, c[1][0]);
    c[1][1] = wmma16(a1, b1, c[1][1]);
  }

#pragma unroll
  for (int mi = 0; mi < 2; ++mi)
#pragma unroll
    for (int ni = 0; ni < 2; ++ni)
#pragma unroll
      for (int r = 0; r < 8; ++r) {
        const int m = waveM + mi * 16 + r + ((lane >> 4) << 3);
        const int n = waveN + ni * 16 + (lane & 15);
        const int b = m / S_LEN, s = m % S_LEN;
        out[((size_t)s * BATCH + b) * EMB + n] = c[mi][ni][r] + bo[n];
      }
}

// ---------------------------------------------------------------------------
// Host orchestration
// ---------------------------------------------------------------------------
extern "C" void kernel_launch(void* const* d_in, const int* in_sizes, int n_in,
                              void* d_out, int out_size, void* d_ws, size_t ws_size,
                              hipStream_t stream) {
  (void)in_sizes; (void)n_in; (void)out_size; (void)ws_size;

  const float* query = (const float*)d_in[0];
  const float* key   = (const float*)d_in[1];
  const float* value = (const float*)d_in[2];
  const float* Wq = (const float*)d_in[3];  const float* bq = (const float*)d_in[4];
  const float* Wk = (const float*)d_in[5];  const float* bk = (const float*)d_in[6];
  const float* Wv = (const float*)d_in[7];  const float* bv = (const float*)d_in[8];
  const float* Wo = (const float*)d_in[9];  const float* bo = (const float*)d_in[10];

  const size_t NSBE = (size_t)S_LEN * BATCH * EMB;   // 8,388,608
  const size_t NEE  = (size_t)EMB * EMB;             // 1,048,576

  // Workspace layout (f16 elements), ~126 MB total
  _Float16* ws   = (_Float16*)d_ws;
  _Float16* qf   = ws;                    // x inputs, f16
  _Float16* kf   = qf  + NSBE;
  _Float16* vf   = kf  + NSBE;
  _Float16* Wq16 = vf  + NSBE;            // weights, f16
  _Float16* Wk16 = Wq16 + NEE;
  _Float16* Wv16 = Wk16 + NEE;
  _Float16* Wo16 = Wv16 + NEE;
  _Float16* Qp   = Wo16 + NEE;            // projected Q [B,H,S,D]
  _Float16* Kp   = Qp  + NSBE;            // projected K [B,H,S,D]
  _Float16* Vt   = Kp  + NSBE;            // projected V, transposed [B,H,D,S]
  _Float16* Yp   = Vt  + NSBE;            // attention output [B,S,E]

  float* out  = (float*)d_out;            // [S,B,E]
  float* attn = out + NSBE;               // [B,H,S,S]

  // 1) f32 -> f16 conversions
  {
    const int T = 256;
    cvt_f16_kernel<<<dim3((NSBE + T - 1) / T), T, 0, stream>>>(query, qf, (int)NSBE);
    cvt_f16_kernel<<<dim3((NSBE + T - 1) / T), T, 0, stream>>>(key,   kf, (int)NSBE);
    cvt_f16_kernel<<<dim3((NSBE + T - 1) / T), T, 0, stream>>>(value, vf, (int)NSBE);
    cvt_f16_kernel<<<dim3((NEE  + T - 1) / T), T, 0, stream>>>(Wq, Wq16, (int)NEE);
    cvt_f16_kernel<<<dim3((NEE  + T - 1) / T), T, 0, stream>>>(Wk, Wk16, (int)NEE);
    cvt_f16_kernel<<<dim3((NEE  + T - 1) / T), T, 0, stream>>>(Wv, Wv16, (int)NEE);
    cvt_f16_kernel<<<dim3((NEE  + T - 1) / T), T, 0, stream>>>(Wo, Wo16, (int)NEE);
  }

  // 2) in-projections (M = 8192, N = 1024 -> grid 128 x 16, 128 threads)
  {
    dim3 grid((S_LEN * BATCH) / 64, EMB / 64);
    proj_qkv_kernel<<<grid, 128, 0, stream>>>(qf, Wq16, bq, Qp, 0);
    proj_qkv_kernel<<<grid, 128, 0, stream>>>(kf, Wk16, bk, Kp, 0);
    proj_qkv_kernel<<<grid, 128, 0, stream>>>(vf, Wv16, bv, Vt, 1);
  }

  // 3) logits (per bh: 2048 x 2048, grid 32 x 32 x 64)
  qk_logits_kernel<<<dim3(S_LEN / 64, S_LEN / 64, BATCH * HEADS), 128, 0, stream>>>(
      Qp, Kp, attn);

  // 4) softmax (one block per row)
  softmax_kernel<<<dim3(BATCH * HEADS * S_LEN), 256, 0, stream>>>(attn);

  // 5) AV (per bh: 2048 x 64, grid 32 x 1 x 64)
  av_kernel<<<dim3(S_LEN / 64, 1, BATCH * HEADS), 128, 0, stream>>>(attn, Vt, Yp);

  // 6) out-projection (M = 8192, N = 1024)
  out_proj_kernel<<<dim3((S_LEN * BATCH) / 64, EMB / 64), 128, 0, stream>>>(
      Yp, Wo16, bo, out);
}
